// CoordinateDecoder_21363167330369
// MI455X (gfx1250) — matrix operand order repaired and checked by reading, not compiled, not run on hardware
//
#include <hip/hip_runtime.h>
#include <hip/hip_bf16.h>

// Problem constants (match reference file).
#define B_   512
#define E_   2048
#define H_   2048
#define G4H_ (4 * H_)
#define T_   256

typedef __attribute__((ext_vector_type(16))) __bf16 v16bf;
typedef __attribute__((ext_vector_type(8)))  __bf16 v8bf;
typedef __attribute__((ext_vector_type(8)))  float  v8f;

__device__ __forceinline__ float sigmoid_f(float x) {
    return 1.0f / (1.0f + __expf(-x));
}

// ---------------------------------------------------------------------------
// Setup kernels
// ---------------------------------------------------------------------------
__global__ __launch_bounds__(256)
void f32_to_bf16_kernel(const float* __restrict__ src, __bf16* __restrict__ dst, int n) {
    int i = blockIdx.x * blockDim.x + threadIdx.x;
    if (i < n) dst[i] = (__bf16)src[i];
}

// zero c state, zero x (prev coord) buffer, and build bias_sum = b_ih + b_hh
__global__ __launch_bounds__(256)
void init_state_kernel(float* __restrict__ c, float* __restrict__ x,
                       const float* __restrict__ b_ih, const float* __restrict__ b_hh,
                       float* __restrict__ bias_sum) {
    int i = blockIdx.x * blockDim.x + threadIdx.x;
    if (i < B_ * H_)  c[i] = 0.0f;
    if (i < B_ * 2)   x[i] = 0.0f;
    if (i < G4H_)     bias_sum[i] = b_ih[i] + b_hh[i];
}

// ---------------------------------------------------------------------------
// bf16 WMMA GEMM:  out[M,N] = A[M,K] @ W[N,K]^T  (+ bias[N]) (+ X[M,2]@Wih[N,2]^T)
//   mode 0: out_f32 = acc + bias ; also write out_bf16  (h0 path)
//   mode 1: out_f32 = acc + bias + X@Wih^T               (gates path)
//
// One wave computes a 32x64 output tile: 2 M-subtiles x 4 N-subtiles = 8
// accumulators of 16x16. Each B fragment (the L2-resident weight stream) is
// reused by 2 WMMAs, halving weight traffic vs a 16x64 tile. All fragments of
// a k-iteration are loaded before the WMMA group so the scheduler can overlap
// loads with the matrix pipe using partial s_wait_loadcnt.
//
// Fragment loads follow the CDNA5 16-bit A(16x32)/B(32x16) VGPR layouts:
//   A: lane L holds row M=L&15, K in [koff..koff+8) U [koff+16..koff+24), koff=(L>=16)*8
//   B: lane L holds col N=L&15, K in [kb..kb+8) U [kb+16..kb+24), kb=(L>=16)*8
// ---------------------------------------------------------------------------
__global__ __launch_bounds__(256)
void gemm_bf16_wmma_kernel(const __bf16* __restrict__ A,    // [M, K]
                           const __bf16* __restrict__ W,    // [N, K]
                           const float*  __restrict__ bias, // [N]
                           const float*  __restrict__ X,    // [M, 2]  (mode 1)
                           const float*  __restrict__ Wih,  // [N, 2]  (mode 1)
                           float*        __restrict__ outF, // [M, N]
                           __bf16*       __restrict__ outBf,// [M, N]  (mode 0)
                           int N, int K, int mode)
{
    const int lane   = threadIdx.x & 31;
    const int wave   = blockIdx.x * (blockDim.x >> 5) + (threadIdx.x >> 5);
    const int tilesN = N >> 6;                // 64-wide N tiles
    const int mtile  = wave / tilesN;         // 32-high M tiles
    const int ntile  = wave - mtile * tilesN;
    const int row0   = mtile << 5;
    const int col0   = ntile << 6;

    const int am   = lane & 15;
    const int akof = (lane >> 4) << 3;        // 0 or 8
    const int bn   = lane & 15;
    const int bkof = (lane >> 4) << 3;        // 0 or 8

    const __bf16* __restrict__ aRow0 = A + (size_t)(row0 + am) * K + akof;
    const __bf16* __restrict__ aRow1 = aRow0 + (size_t)16 * K;
    const __bf16* __restrict__ bCol0 = W + (size_t)(col0 + bn) * K + bkof;

    v8f acc0[4] = {};   // rows row0 .. row0+15
    v8f acc1[4] = {};   // rows row0+16 .. row0+31

    union Frag { v16bf v; v8bf h[2]; };

    for (int kk = 0; kk < K; kk += 32) {
        // ---- load all fragments for this k-slice first ----
        Frag a0, a1;
        a0.h[0] = *(const v8bf*)(aRow0 + kk);
        a0.h[1] = *(const v8bf*)(aRow0 + kk + 16);
        a1.h[0] = *(const v8bf*)(aRow1 + kk);
        a1.h[1] = *(const v8bf*)(aRow1 + kk + 16);

        Frag b[4];
#pragma unroll
        for (int nt = 0; nt < 4; ++nt) {
            const __bf16* bp = bCol0 + (size_t)nt * 16 * K + kk;
            b[nt].h[0] = *(const v8bf*)(bp);
            b[nt].h[1] = *(const v8bf*)(bp + 16);
        }

        // Prefetch next K-slice (global_prefetch_b8) to hide L2 latency.
        if (kk + 32 < K) {
            __builtin_prefetch(aRow0 + kk + 32, 0, 1);
            __builtin_prefetch(bCol0 + kk + 32, 0, 1);
        }

        // ---- 8 WMMAs: each B fragment feeds both M-subtiles ----
#pragma unroll
        for (int nt = 0; nt < 4; ++nt) {
            acc0[nt] = __builtin_amdgcn_wmma_f32_16x16x32_bf16(
                false, a0.v, false, b[nt].v, (short)0, acc0[nt], false, false);
            acc1[nt] = __builtin_amdgcn_wmma_f32_16x16x32_bf16(
                false, a1.v, false, b[nt].v, (short)0, acc1[nt], false, false);
        }
    }

    // Epilogue. C/D layout: VGPR r, lanes 0-15 -> M=r, lanes 16-31 -> M=8+r,
    // N = lane & 15 within the 16-wide sub-tile.
    const int rbase = row0 + ((lane >> 4) << 3);
#pragma unroll
    for (int nt = 0; nt < 4; ++nt) {
        const int col = col0 + nt * 16 + (lane & 15);
        const float bv = bias[col];
        float w0 = 0.0f, w1 = 0.0f;
        if (mode == 1) { w0 = Wih[col * 2 + 0]; w1 = Wih[col * 2 + 1]; }
#pragma unroll
        for (int half = 0; half < 2; ++half) {
            const v8f* acc = half ? acc1 : acc0;
#pragma unroll
            for (int r = 0; r < 8; ++r) {
                const int row = rbase + half * 16 + r;
                float v = acc[nt][r] + bv;
                if (mode == 1)
                    v += X[row * 2 + 0] * w0 + X[row * 2 + 1] * w1;
                outF[(size_t)row * N + col] = v;
                if (mode == 0)
                    outBf[(size_t)row * N + col] = (__bf16)v;
            }
        }
    }
}

// ---------------------------------------------------------------------------
// Pointwise LSTM cell: gates (B,4H) -> updated c, h (f32 + bf16 copies)
// torch gate order i, f, g, o
// ---------------------------------------------------------------------------
__global__ __launch_bounds__(256)
void lstm_cell_kernel(const float* __restrict__ gates,
                      float* __restrict__ c,
                      float* __restrict__ hF,
                      __bf16* __restrict__ hBf)
{
    const int idx = blockIdx.x * blockDim.x + threadIdx.x;   // 0 .. B*H-1
    const int b = idx >> 11;            // / H_
    const int j = idx & (H_ - 1);
    const float* g = gates + (size_t)b * G4H_;
    const float ig = sigmoid_f(g[j]);
    const float fg = sigmoid_f(g[H_ + j]);
    const float gg = tanhf(g[2 * H_ + j]);
    const float og = sigmoid_f(g[3 * H_ + j]);
    const float cn = fg * c[idx] + ig * gg;
    const float hn = og * tanhf(cn);
    c[idx]  = cn;
    hF[idx] = hn;
    hBf[idx] = (__bf16)hn;
}

// ---------------------------------------------------------------------------
// Output head: per batch row, 3 dot products of length H (coord0, coord1, stop).
// One wave per batch row; butterfly reduction across the 32 lanes.
// Writes coord into X (next step's input) and into d_out; stop into d_out.
// ---------------------------------------------------------------------------
__global__ __launch_bounds__(256)
void head_kernel(const float* __restrict__ hF,
                 const float* __restrict__ Wout,  // [2, H]
                 const float* __restrict__ bout,  // [2]
                 const float* __restrict__ Wstop, // [1, H]
                 const float* __restrict__ bstop, // [1]
                 float* __restrict__ X,           // [B, 2]
                 float* __restrict__ outC,        // [B, T, 2]
                 float* __restrict__ outS,        // [B, T, 1]
                 int t)
{
    const int lane = threadIdx.x & 31;
    const int b    = blockIdx.x * (blockDim.x >> 5) + (threadIdx.x >> 5);
    const float* h = hF + (size_t)b * H_;

    float s0 = 0.0f, s1 = 0.0f, s2 = 0.0f;
    for (int k = lane; k < H_; k += 32) {
        const float hv = h[k];
        s0 += hv * Wout[k];
        s1 += hv * Wout[H_ + k];
        s2 += hv * Wstop[k];
    }
#pragma unroll
    for (int off = 16; off > 0; off >>= 1) {
        s0 += __shfl_xor(s0, off, 32);
        s1 += __shfl_xor(s1, off, 32);
        s2 += __shfl_xor(s2, off, 32);
    }
    if (lane == 0) {
        const float c0 = s0 + bout[0];
        const float c1 = s1 + bout[1];
        const float sp = sigmoid_f(s2 + bstop[0]);
        X[b * 2 + 0] = c0;
        X[b * 2 + 1] = c1;
        const size_t o = ((size_t)b * T_ + t) * 2;
        outC[o + 0] = c0;
        outC[o + 1] = c1;
        outS[(size_t)b * T_ + t] = sp;
    }
}

// ---------------------------------------------------------------------------
// Launch
// ---------------------------------------------------------------------------
extern "C" void kernel_launch(void* const* d_in, const int* in_sizes, int n_in,
                              void* d_out, int out_size, void* d_ws, size_t ws_size,
                              hipStream_t stream)
{
    const float* embedding = (const float*)d_in[0];
    // d_in[1] = max_seq_length (== 256 per reference constants)
    const float* W_embed = (const float*)d_in[2];
    const float* b_embed = (const float*)d_in[3];
    const float* W_ih    = (const float*)d_in[4];
    const float* b_ih    = (const float*)d_in[5];
    const float* W_hh    = (const float*)d_in[6];
    const float* b_hh    = (const float*)d_in[7];
    const float* W_out   = (const float*)d_in[8];
    const float* b_out   = (const float*)d_in[9];
    const float* W_stop  = (const float*)d_in[10];
    const float* b_stop  = (const float*)d_in[11];

    // Workspace layout (~71 MB total)
    char* ws = (char*)d_ws;
    __bf16* Whh_bf  = (__bf16*)ws;  ws += (size_t)G4H_ * H_ * sizeof(__bf16); // 33.5 MB (L2-resident)
    __bf16* Wemb_bf = (__bf16*)ws;  ws += (size_t)H_ * E_ * sizeof(__bf16);   //  8.4 MB
    __bf16* emb_bf  = (__bf16*)ws;  ws += (size_t)B_ * E_ * sizeof(__bf16);   //  2.1 MB
    __bf16* h_bf    = (__bf16*)ws;  ws += (size_t)B_ * H_ * sizeof(__bf16);   //  2.1 MB
    float*  h_f     = (float*)ws;   ws += (size_t)B_ * H_ * sizeof(float);    //  4.2 MB
    float*  c_f     = (float*)ws;   ws += (size_t)B_ * H_ * sizeof(float);    //  4.2 MB
    float*  gates   = (float*)ws;   ws += (size_t)B_ * G4H_ * sizeof(float);  // 16.8 MB
    float*  bias_s  = (float*)ws;   ws += (size_t)G4H_ * sizeof(float);
    float*  xbuf    = (float*)ws;   ws += (size_t)B_ * 2 * sizeof(float);

    float* outC = (float*)d_out;                    // (B, T, 2)
    float* outS = outC + (size_t)B_ * T_ * 2;       // (B, T, 1)

    // --- one-time setup (re-run every call: deterministic) ---
    f32_to_bf16_kernel<<<(G4H_ * H_) / 256, 256, 0, stream>>>(W_hh, Whh_bf, G4H_ * H_);
    f32_to_bf16_kernel<<<(H_ * E_) / 256, 256, 0, stream>>>(W_embed, Wemb_bf, H_ * E_);
    f32_to_bf16_kernel<<<(B_ * E_) / 256, 256, 0, stream>>>(embedding, emb_bf, B_ * E_);
    init_state_kernel<<<(B_ * H_) / 256, 256, 0, stream>>>(c_f, xbuf, b_ih, b_hh, bias_s);

    // --- h0 = embedding @ W_embed^T + b_embed ---
    {
        const int waves = (B_ / 32) * (H_ / 64);    // 16*32 = 512 waves
        gemm_bf16_wmma_kernel<<<waves / 8, 256, 0, stream>>>(
            emb_bf, Wemb_bf, b_embed, nullptr, nullptr,
            h_f, h_bf, H_, E_, /*mode=*/0);
    }

    // --- 256 sequential LSTM steps ---
    const int gw = (B_ / 32) * (G4H_ / 64);         // 16*128 = 2048 waves per step GEMM
    for (int t = 0; t < T_; ++t) {
        gemm_bf16_wmma_kernel<<<gw / 8, 256, 0, stream>>>(
            h_bf, Whh_bf, bias_s, xbuf, W_ih,
            gates, nullptr, G4H_, H_, /*mode=*/1);
        lstm_cell_kernel<<<(B_ * H_) / 256, 256, 0, stream>>>(gates, c_f, h_f, h_bf);
        head_kernel<<<B_ / 8, 256, 0, stream>>>(h_f, W_out, b_out, W_stop, b_stop,
                                                xbuf, outC, outS, t);
    }
}
